// GSA_19567871001251
// MI455X (gfx1250) — matrix operand order repaired and checked by reading, not compiled, not run on hardware
//
#include <hip/hip_runtime.h>

// ---------------------------------------------------------------------------
// Problem constants (from reference)
// ---------------------------------------------------------------------------
#define BB 8
#define TT 1024
#define DIMF 512
#define HH 8
#define DKK 64
#define DHH 512
#define LL 32
#define O3 1536   // 3*DH

typedef __attribute__((ext_vector_type(16))) __bf16 v16bf;
typedef __attribute__((ext_vector_type(8)))  float  v8f;

// ---------------------------------------------------------------------------
// WMMA helpers (CDNA5 v_wmma_f32_16x16x32_bf16, wave32)
// A 16x32 (MxK): lane L -> row m=L&15; elem j -> K = j + 16*(L>>4)
// B 32x16 (KxN): lane L -> col n=L&15; elem j -> K = j + 16*(L>>4)
// C/D 16x16:     lane L -> n=L&15; vgpr r -> m = r + 8*(L>>4)
// ---------------------------------------------------------------------------
__device__ __forceinline__ v8f wmma_bf16(v16bf a, v16bf b, v8f c) {
    return __builtin_amdgcn_wmma_f32_16x16x32_bf16(false, a, false, b, (short)0, c,
                                                   false, false);
}

// A fragment from a row-major f32 row pointer (rowk = &A[m][kb])
__device__ __forceinline__ v16bf a_frag_f32(const float* rowk, int half) {
    v16bf r;
    const float* p0 = rowk + 16 * half;       // j = 0..7
    const float* p1 = rowk + 16 * half + 8;   // j = 8..15
#pragma unroll
    for (int j = 0; j < 8; ++j) {
        r[j]     = (__bf16)p0[j];
        r[8 + j] = (__bf16)p1[j];
    }
    return r;
}

// B fragment from 16 contiguous f32 (p = &B[kb + 16*half][n] in K-contiguous layout)
__device__ __forceinline__ v16bf b_frag_f32(const float* p) {
    v16bf r;
#pragma unroll
    for (int j = 0; j < 16; ++j) r[j] = (__bf16)p[j];
    return r;
}

// B fragment from 16 contiguous bf16
__device__ __forceinline__ v16bf b_frag_bf16(const __bf16* p) {
    v16bf r;
#pragma unroll
    for (int j = 0; j < 16; ++j) r[j] = p[j];
    return r;
}

// ---------------------------------------------------------------------------
// Kernel 1: qkv[b][o][t] = sum_f Wqkv[o][f] * x[b][t][f]
// A = Wqkv (row-major, K contiguous). B[k=f][n=t] = x[b][t][f] (K contiguous).
// One wave -> 32(M) x 64(N) strip. All 4 B fragments get distinct registers
// per k-step so the bf16 WMMA->VALU WAR hazard is covered by real work, not
// v_nops. 768 blocks x 256 thr: 8b * 48 m-groups * 16 n-groups = 6144 waves.
// ---------------------------------------------------------------------------
__global__ void k_qkv(const float* __restrict__ x, const float* __restrict__ W,
                      float* __restrict__ qkv) {
    const int w    = blockIdx.x * 8 + (threadIdx.x >> 5);
    const int b    = w / 768;
    const int rem  = w % 768;
    const int m0   = (rem / 16) * 32;   // 48 m-groups of 32 rows
    const int t0   = (rem % 16) * 64;   // 16 n-groups of 64
    const int lane = threadIdx.x & 31;
    const int nl   = lane & 15;
    const int half = lane >> 4;

    const float* Arow0 = W + (size_t)(m0 + nl) * DIMF;
    const float* Arow1 = Arow0 + (size_t)16 * DIMF;
    const float* xb    = x + (size_t)b * TT * DIMF;

    v8f acc[2][4] = {{{}, {}, {}, {}}, {{}, {}, {}, {}}};
    for (int kb = 0; kb < DIMF; kb += 32) {
        const v16bf a0 = a_frag_f32(Arow0 + kb, half);
        const v16bf a1 = a_frag_f32(Arow1 + kb, half);
        v16bf bfv[4];
#pragma unroll
        for (int nt = 0; nt < 4; ++nt)
            bfv[nt] = b_frag_f32(xb + (size_t)(t0 + nt * 16 + nl) * DIMF +
                                 kb + 16 * half);
#pragma unroll
        for (int nt = 0; nt < 4; ++nt) {
            acc[0][nt] = wmma_bf16(a0, bfv[nt], acc[0][nt]);
            acc[1][nt] = wmma_bf16(a1, bfv[nt], acc[1][nt]);
        }
    }
    float* ob = qkv + (size_t)b * O3 * TT;
#pragma unroll
    for (int mt = 0; mt < 2; ++mt)
#pragma unroll
        for (int nt = 0; nt < 4; ++nt)
#pragma unroll
            for (int r = 0; r < 8; ++r)
                ob[(size_t)(m0 + mt * 16 + r + 8 * half) * TT +
                   (t0 + nt * 16 + nl)] = acc[mt][nt][r];
}

// ---------------------------------------------------------------------------
// Kernel 2: in-place softmax over t for k rows (qkv rows [512,1024) per batch).
// One block per row; deterministic tree reductions.
// ---------------------------------------------------------------------------
__global__ void k_softmax(float* __restrict__ qkv) {
    __shared__ float red[256];
    const int blk = blockIdx.x;          // 4096 = 8*512
    const int b   = blk >> 9;
    const int o   = 512 + (blk & 511);
    float* row    = qkv + (size_t)b * O3 * TT + (size_t)o * TT;
    const int tid = threadIdx.x;

    float v[4];
#pragma unroll
    for (int i = 0; i < 4; ++i) v[i] = row[tid + i * 256];
    float m = fmaxf(fmaxf(v[0], v[1]), fmaxf(v[2], v[3]));
    red[tid] = m;
    __syncthreads();
    for (int s = 128; s > 0; s >>= 1) {
        if (tid < s) red[tid] = fmaxf(red[tid], red[tid + s]);
        __syncthreads();
    }
    m = red[0];
    __syncthreads();
    float e[4], ls = 0.f;
#pragma unroll
    for (int i = 0; i < 4; ++i) { e[i] = __expf(v[i] - m); ls += e[i]; }
    red[tid] = ls;
    __syncthreads();
    for (int s = 128; s > 0; s >>= 1) {
        if (tid < s) red[tid] += red[tid + s];
        __syncthreads();
    }
    const float inv = 1.0f / red[0];
#pragma unroll
    for (int i = 0; i < 4; ++i) row[tid + i * 256] = e[i] * inv;
}

// ---------------------------------------------------------------------------
// Kernel 3: Psum[t][d] = sum over valid band offsets of rel_positions[r][d]
// valid: 0 <= t + r - 31 < T  ->  r in [max(0,31-t), min(62, 1054-t)]
// ---------------------------------------------------------------------------
__global__ void k_psum(const float* __restrict__ relpos, float* __restrict__ psum) {
    const int t = blockIdx.x;
    const int d = threadIdx.x;
    int rlo = 31 - t;    if (rlo < 0)  rlo = 0;
    int rhi = 1054 - t;  if (rhi > 62) rhi = 62;
    float s = 0.f;
    for (int r = rlo; r <= rhi; ++r) s += relpos[r * DKK + d];
    psum[t * DKK + d] = s;
}

// ---------------------------------------------------------------------------
// Kernel 4: per head n (64 blocks x 256 thr):
//  stage1: context[d][e] = sum_t ksm[d][t] v[e][t]      (WMMA, K=1024)
//  stage2: content[e][t] = sum_d ctx[d][e] q[d][t]      (WMMA, K=64)
//  stage3: S[t] = sum_d q[d][t] Psum[t][d]; rel = v*S; BN partial sums per e
// ---------------------------------------------------------------------------
__global__ void k_head(const float* __restrict__ qkv, const float* __restrict__ psum,
                       float* __restrict__ content, float* __restrict__ rel,
                       float* __restrict__ bnpart) {
    __shared__ float ctx[64 * 64];
    __shared__ float Sld[1024];
    __shared__ float part[512];   // [e][j] sums then [e][j] sumsq: 64*4*2

    const int n = blockIdx.x;
    const int b = n >> 3, h = n & 7;
    const float* qbase = qkv + (size_t)b * O3 * TT + (size_t)(h * DKK) * TT;
    const float* kbase = qbase + (size_t)512 * TT;
    const float* vbase = qbase + (size_t)1024 * TT;

    const int wv   = threadIdx.x >> 5;
    const int lane = threadIdx.x & 31;
    const int nl   = lane & 15;
    const int half = lane >> 4;

    // ---- stage 1: context (16 tiles of 16x16 over (d,e); 2 tiles per wave) ----
    // K unrolled by 2 with distinct fragment registers to avoid WAR v_nops.
    for (int tt = 2 * wv; tt < 2 * wv + 2; ++tt) {
        const int mtile = (tt >> 2) * 16;   // d
        const int ntile = (tt & 3) * 16;    // e
        const float* Arow = kbase + (size_t)(mtile + nl) * TT;
        const float* Bcol = vbase + (size_t)(ntile + nl) * TT;
        v8f acc = {};
        for (int kb = 0; kb < TT; kb += 64) {
            const v16bf a0 = a_frag_f32(Arow + kb, half);
            const v16bf b0 = b_frag_f32(Bcol + kb + 16 * half);
            const v16bf a1 = a_frag_f32(Arow + kb + 32, half);
            const v16bf b1 = b_frag_f32(Bcol + kb + 32 + 16 * half);
            acc = wmma_bf16(a0, b0, acc);
            acc = wmma_bf16(a1, b1, acc);
        }
#pragma unroll
        for (int r = 0; r < 8; ++r)
            ctx[(mtile + r + 8 * half) * 64 + ntile + nl] = acc[r];
    }
    __syncthreads();

    // ---- stage 2: content = ctx^T x Q ; wave handles 128-wide t chunk ----
    // The 8 A fragments (4 m-tiles x 2 k-steps) are invariant across the nt
    // loop: build them once from LDS (saves 8x LDS traffic + cvt work, and
    // they are never overwritten -> no WMMA WAR hazard on A).
    const int tb0 = wv * 128;
    v16bf afv[8];
#pragma unroll
    for (int ks2 = 0; ks2 < 2; ++ks2)
#pragma unroll
        for (int mt = 0; mt < 4; ++mt) {
            v16bf af;
#pragma unroll
            for (int j = 0; j < 16; ++j) {
                const int kk = ks2 * 32 + 16 * half + j;
                af[j] = (__bf16)ctx[kk * 64 + mt * 16 + nl];
            }
            afv[ks2 * 4 + mt] = af;
        }

    for (int nt = 0; nt < 8; ++nt) {
        const int tcol = tb0 + nt * 16 + nl;
        v8f acc[4] = {{}, {}, {}, {}};
        v16bf bf0, bf1;                     // B[k=d][n=t] = q[d][t] (stride T)
        const float* qp0 = qbase + (size_t)(16 * half) * TT + tcol;
        const float* qp1 = qbase + (size_t)(32 + 16 * half) * TT + tcol;
#pragma unroll
        for (int j = 0; j < 16; ++j) bf0[j] = (__bf16)qp0[(size_t)j * TT];
#pragma unroll
        for (int j = 0; j < 16; ++j) bf1[j] = (__bf16)qp1[(size_t)j * TT];
#pragma unroll
        for (int mt = 0; mt < 4; ++mt) acc[mt] = wmma_bf16(afv[mt], bf0, acc[mt]);
#pragma unroll
        for (int mt = 0; mt < 4; ++mt) acc[mt] = wmma_bf16(afv[4 + mt], bf1, acc[mt]);
#pragma unroll
        for (int mt = 0; mt < 4; ++mt)
#pragma unroll
            for (int r = 0; r < 8; ++r)
                content[(size_t)(n * 64 + mt * 16 + r + 8 * half) * TT +
                        tb0 + nt * 16 + nl] = acc[mt][r];
    }
    __syncthreads();

    // ---- stage 3a: S[t] ----
    for (int t = threadIdx.x; t < TT; t += 256) {
        float s = 0.f;
#pragma unroll 8
        for (int d = 0; d < DKK; ++d) s += qbase[(size_t)d * TT + t] * psum[t * DKK + d];
        Sld[t] = s;
    }
    __syncthreads();

    // ---- stage 3b: rel = v*S, deterministic BN partials (4 threads per e) ----
    const int e = threadIdx.x >> 2;
    const int j = threadIdx.x & 3;
    const float* vrow = vbase + (size_t)e * TT;
    float* relrow = rel + (size_t)(n * 64 + e) * TT;
    float sm = 0.f, sq = 0.f;
    for (int t = j * 256; t < j * 256 + 256; ++t) {
        const float r = vrow[t] * Sld[t];
        relrow[t] = r;
        sm += r;
        sq += r * r;
    }
    part[e * 4 + j]       = sm;
    part[256 + e * 4 + j] = sq;
    __syncthreads();
    if (j == 0) {
        float s0 = part[e * 4] + part[e * 4 + 1] + part[e * 4 + 2] + part[e * 4 + 3];
        float s1 = part[256 + e * 4] + part[256 + e * 4 + 1] +
                   part[256 + e * 4 + 2] + part[256 + e * 4 + 3];
        bnpart[(n * 64 + e) * 2]     = s0;
        bnpart[(n * 64 + e) * 2 + 1] = s1;
    }
}

// ---------------------------------------------------------------------------
// Kernel 5: BN finalize (1 block, 64 threads), fixed-order sum over n
// bncoef[e] = gamma*invstd, bncoef[64+e] = beta - mu*gamma*invstd
// ---------------------------------------------------------------------------
__global__ void k_bnfin(const float* __restrict__ bnpart, const float* __restrict__ gamma,
                        const float* __restrict__ beta, float* __restrict__ bncoef) {
    const int e = threadIdx.x;
    float sm = 0.f, sq = 0.f;
    for (int nn = 0; nn < 64; ++nn) {
        sm += bnpart[(nn * 64 + e) * 2];
        sq += bnpart[(nn * 64 + e) * 2 + 1];
    }
    const float cnt = 1.0f / 65536.0f;
    const float mu  = sm * cnt;
    float var = sq * cnt - mu * mu;
    var = fmaxf(var, 0.f);
    const float inv = rsqrtf(var + 1e-5f);
    const float sc  = inv * gamma[e];
    bncoef[e]      = sc;
    bncoef[64 + e] = beta[e] - mu * sc;
}

// ---------------------------------------------------------------------------
// Kernel 6: hidden = content + BN(rel); transpose (b,dh,t)->(b,t,dh) as bf16
// so the final GEMM gets K-contiguous B fragments. 32x32 LDS tile transpose.
// ---------------------------------------------------------------------------
__global__ void k_fuse(const float* __restrict__ content, const float* __restrict__ rel,
                       const float* __restrict__ bncoef, __bf16* __restrict__ hbf) {
    __shared__ float tile[32][33];
    const int blk = blockIdx.x;           // 8 * 32 * 16
    const int b   = blk >> 9;
    const int rem = blk & 511;
    const int t0  = (rem >> 4) * 32;
    const int d0  = (rem & 15) * 32;

#pragma unroll
    for (int it = 0; it < 4; ++it) {
        const int lin = threadIdx.x + it * 256;
        const int ty  = lin >> 5;         // dh offset
        const int tx  = lin & 31;         // t offset (coalesced read)
        const int dh  = d0 + ty;
        const int e   = dh & 63;
        const size_t idx = (size_t)(b * DHH + dh) * TT + t0 + tx;
        tile[ty][tx] = content[idx] + rel[idx] * bncoef[e] + bncoef[64 + e];
    }
    __syncthreads();
#pragma unroll
    for (int it = 0; it < 4; ++it) {
        const int lin = threadIdx.x + it * 256;
        const int tw  = lin >> 5;         // t offset
        const int dw  = lin & 31;         // dh offset (coalesced write)
        hbf[(size_t)(b * TT + t0 + tw) * DHH + d0 + dw] = (__bf16)tile[dw][tw];
    }
}

// ---------------------------------------------------------------------------
// Kernel 7: out[b][o][t] = sum_dh Wout[o][dh] * hidden[b][t][dh] + bout[o]
// 32(M) x 64(N) per wave like k_qkv; B operand already bf16, K-contiguous.
// 256 blocks x 256 thr: 8b * 16 m-groups * 16 n-groups = 2048 waves.
// ---------------------------------------------------------------------------
__global__ void k_out(const float* __restrict__ Wout, const float* __restrict__ bout,
                      const __bf16* __restrict__ hbf, float* __restrict__ out) {
    const int w    = blockIdx.x * 8 + (threadIdx.x >> 5);
    const int b    = w >> 8;
    const int rem  = w & 255;
    const int m0   = (rem >> 4) * 32;  // 16 m-groups of 32 rows
    const int t0   = (rem & 15) * 64;  // 16 n-groups
    const int lane = threadIdx.x & 31;
    const int nl   = lane & 15;
    const int half = lane >> 4;

    const float*  Arow0 = Wout + (size_t)(m0 + nl) * DHH;
    const float*  Arow1 = Arow0 + (size_t)16 * DHH;
    const __bf16* hb    = hbf + (size_t)b * TT * DHH;

    v8f acc[2][4] = {{{}, {}, {}, {}}, {{}, {}, {}, {}}};
    for (int kb = 0; kb < DHH; kb += 32) {
        const v16bf a0 = a_frag_f32(Arow0 + kb, half);
        const v16bf a1 = a_frag_f32(Arow1 + kb, half);
        v16bf bfv[4];
#pragma unroll
        for (int nt = 0; nt < 4; ++nt)
            bfv[nt] = b_frag_bf16(hb + (size_t)(t0 + nt * 16 + nl) * DHH +
                                  kb + 16 * half);
#pragma unroll
        for (int nt = 0; nt < 4; ++nt) {
            acc[0][nt] = wmma_bf16(a0, bfv[nt], acc[0][nt]);
            acc[1][nt] = wmma_bf16(a1, bfv[nt], acc[1][nt]);
        }
    }
#pragma unroll
    for (int mt = 0; mt < 2; ++mt)
#pragma unroll
        for (int nt = 0; nt < 4; ++nt)
#pragma unroll
            for (int r = 0; r < 8; ++r) {
                const int m = m0 + mt * 16 + r + 8 * half;
                out[(size_t)(b * DHH + m) * TT + t0 + nt * 16 + nl] =
                    acc[mt][nt][r] + bout[m];
            }
}

// ---------------------------------------------------------------------------
extern "C" void kernel_launch(void* const* d_in, const int* in_sizes, int n_in,
                              void* d_out, int out_size, void* d_ws, size_t ws_size,
                              hipStream_t stream) {
    (void)in_sizes; (void)n_in; (void)out_size; (void)ws_size;
    const float* x     = (const float*)d_in[0];
    const float* Wqkv  = (const float*)d_in[1];
    const float* Wout  = (const float*)d_in[2];
    const float* bout  = (const float*)d_in[3];
    const float* relp  = (const float*)d_in[4];
    const float* gamma = (const float*)d_in[5];
    const float* beta  = (const float*)d_in[6];
    float* out = (float*)d_out;

    // Workspace carve-up (all fully written before read; no cross-call state)
    float* ws      = (float*)d_ws;
    float* qkv     = ws;                       // 8*1536*1024      = 12,582,912 f32
    float* psum    = qkv + 12582912;           // 1024*64          =     65,536 f32
    float* content = psum + 65536;             // 8*512*1024       =  4,194,304 f32
    float* rel     = content + 4194304;        // 8*512*1024       =  4,194,304 f32
    float* bnpart  = rel + 4194304;            // 64*64*2          =      8,192 f32
    float* bncoef  = bnpart + 8192;            // 128 f32
    __bf16* hbf    = (__bf16*)(bncoef + 128);  // 8*1024*512 bf16

    k_qkv    <<< 768, 256, 0, stream>>>(x, Wqkv, qkv);
    k_softmax<<<4096, 256, 0, stream>>>(qkv);
    k_psum   <<<1024,  64, 0, stream>>>(relp, psum);
    k_head   <<<  64, 256, 0, stream>>>(qkv, psum, content, rel, bnpart);
    k_bnfin  <<<   1,  64, 0, stream>>>(bnpart, gamma, beta, bncoef);
    k_fuse   <<<4096, 256, 0, stream>>>(content, rel, bncoef, hbf);
    k_out    <<< 256, 256, 0, stream>>>(Wout, bout, hbf, out);
}
